// GraphAttentionLayer_41266045780241
// MI455X (gfx1250) — matrix operand order repaired and checked by reading, not compiled, not run on hardware
//
#include <hip/hip_runtime.h>
#include <hip/hip_bf16.h>
#include <stdint.h>

typedef __attribute__((ext_vector_type(16))) __bf16 v16bf;
typedef __attribute__((ext_vector_type(8)))  __bf16 v8bf;
typedef __attribute__((ext_vector_type(8)))  float  v8f;
typedef __attribute__((ext_vector_type(4)))  int    v4i;

typedef __attribute__((address_space(1))) v4i gas_v4i;   // global AS pointer-to-v4i
typedef __attribute__((address_space(3))) v4i las_v4i;   // LDS    AS pointer-to-v4i

#define IN_F   256
#define OUT_F  128
#define HEADS  4
#define NNODES 2048
#define NB     8
#define ALPHA_LRELU 0.2f
#define NEG_BIG (-9.0e15f)

#if __has_builtin(__builtin_amdgcn_global_load_async_to_lds_b128)
#define HAS_ASYNC_LDS 1
#else
#define HAS_ASYNC_LDS 0
#endif

// ---------------- conversion kernels ----------------
__global__ void cvt_h_kernel(const float* __restrict__ h, __bf16* __restrict__ hB, int n) {
  int i = blockIdx.x * blockDim.x + threadIdx.x;
  int stride = gridDim.x * blockDim.x;
  for (; i < n; i += stride) hB[i] = (__bf16)h[i];
}

// WtB[h][o][k] = W[h][k][o]  (bf16)
__global__ void cvt_w_kernel(const float* __restrict__ W, __bf16* __restrict__ WtB) {
  int i = blockIdx.x * blockDim.x + threadIdx.x;   // HEADS*OUT_F*IN_F = 131072
  int hh = i >> 15;
  int o  = (i >> 8) & (OUT_F - 1);
  int k  = i & (IN_F - 1);
  WtB[i] = (__bf16)W[((hh << 8) + k) * OUT_F + o];
}

// ---------------- GEMM1: Wh = h @ W, stored transposed WhT[b][h][o][n] bf16 ----------------
__global__ void __launch_bounds__(256) gemm1_kernel(const __bf16* __restrict__ hB,
                                                    const __bf16* __restrict__ WtB,
                                                    __bf16* __restrict__ WhT) {
  int wid  = blockIdx.x * 8 + (threadIdx.x >> 5);   // 4096 waves
  int lane = threadIdx.x & 31;
  int lo = lane & 15, hi = lane >> 4;
  int ntile = wid & 127;
  int bh    = wid >> 7;                 // 0..31
  int b = bh >> 2, hh = bh & 3;
  int n0 = ntile << 4;

  const __bf16* Arow = hB + (size_t)(b * NNODES + n0 + lo) * IN_F;
  v8f acc[8] = {};

  for (int k0 = 0; k0 < IN_F; k0 += 32) {
    v8bf a0 = *(const v8bf*)(Arow + k0 + 16 * hi);
    v8bf a1 = *(const v8bf*)(Arow + k0 + 16 * hi + 8);
    v16bf A = __builtin_shufflevector(a0, a1, 0,1,2,3,4,5,6,7,8,9,10,11,12,13,14,15);
#pragma unroll
    for (int j = 0; j < 8; ++j) {
      const __bf16* Bp = WtB + (size_t)(hh * OUT_F + (j << 4) + lo) * IN_F + k0 + 16 * hi;
      v16bf Bv = *(const v16bf*)Bp;
      acc[j] = __builtin_amdgcn_wmma_f32_16x16x32_bf16(false, A, false, Bv,
                                                       (short)0, acc[j], false, false);
    }
  }
#pragma unroll
  for (int j = 0; j < 8; ++j) {
    v8bf p;
#pragma unroll
    for (int r = 0; r < 8; ++r) p[r] = (__bf16)acc[j][r];
    __bf16* dst = WhT + (size_t)(bh * OUT_F + (j << 4) + lo) * NNODES + n0 + 8 * hi;
    *(v8bf*)dst = p;   // lane's 8 rows are consecutive n -> one packed 16B store
  }
}

// ---------------- s1/s2 = Wh . a1 / a2 ----------------
__global__ void s12_kernel(const __bf16* __restrict__ WhT, const float* __restrict__ a,
                           float* __restrict__ s1, float* __restrict__ s2) {
  int idx = blockIdx.x * blockDim.x + threadIdx.x;  // 65536 = B*H*N
  int bh = idx >> 11, n = idx & (NNODES - 1);
  int hh = bh & 3;
  const __bf16* base = WhT + (size_t)bh * OUT_F * NNODES + n;
  const float* a1 = a + hh * 2 * OUT_F;
  const float* a2 = a1 + OUT_F;
  float acc1 = 0.f, acc2 = 0.f;
  for (int o = 0; o < OUT_F; ++o) {
    float w = (float)base[(size_t)o * NNODES];
    acc1 += w * a1[o];
    acc2 += w * a2[o];
  }
  s1[idx] = acc1; s2[idx] = acc2;
}

// ---------------- per-row softmax stats: rowC = max + ln(sum exp(x-max)) ----------------
__global__ void __launch_bounds__(256) stats_kernel(const float* __restrict__ s1,
                                                    const float* __restrict__ s2,
                                                    const int* __restrict__ adj,
                                                    float* __restrict__ rowC) {
  int row  = blockIdx.x * 8 + (threadIdx.x >> 5);   // 65536 rows
  int lane = threadIdx.x & 31;
  int bh = row >> 11, n = row & (NNODES - 1);
  float s1v = s1[row];
  const int*   adjRow = adj + (size_t)n * NNODES;
  const float* s2row  = s2 + (size_t)bh * NNODES;

  float mx = NEG_BIG;
  for (int m = lane; m < NNODES; m += 32) {
    float x = s1v + s2row[m];
    x = fmaxf(x, ALPHA_LRELU * x);
    x = adjRow[m] > 0 ? x : NEG_BIG;
    mx = fmaxf(mx, x);
  }
#pragma unroll
  for (int off = 16; off > 0; off >>= 1) mx = fmaxf(mx, __shfl_xor(mx, off, 32));

  float sum = 0.f;
  for (int m = lane; m < NNODES; m += 32) {
    float x = s1v + s2row[m];
    x = fmaxf(x, ALPHA_LRELU * x);
    x = adjRow[m] > 0 ? x : NEG_BIG;
    sum += __expf(x - mx);
  }
#pragma unroll
  for (int off = 16; off > 0; off >>= 1) sum += __shfl_xor(sum, off, 32);

  if (lane == 0) rowC[row] = mx + logf(sum);
}

// ---------------- fused attention GEMM + head-mean + ELU ----------------
__device__ __forceinline__ __bf16 pval(float sv, float c, int ai, float s2v) {
  float x = sv + s2v;
  x = fmaxf(x, ALPHA_LRELU * x);
  x = ai > 0 ? x : NEG_BIG;
  return (__bf16)__expf(x - c);   // masked -> exp(very negative) = 0, matches reference
}

// 16B global -> LDS copy: async (bypasses VGPRs, ASYNCcnt-tracked) when available
__device__ __forceinline__ void copy16_to_lds(const void* g, void* l) {
#if HAS_ASYNC_LDS
  __builtin_amdgcn_global_load_async_to_lds_b128(
      (gas_v4i*)(uintptr_t)g, (las_v4i*)(uintptr_t)l, 0, 0);
#else
  *(int4*)l = *(const int4*)g;
#endif
}

__device__ __forceinline__ void async_wait_all() {
#if HAS_ASYNC_LDS
#if __has_builtin(__builtin_amdgcn_s_wait_asynccnt)
  __builtin_amdgcn_s_wait_asynccnt(0);
#else
  asm volatile("s_wait_asynccnt 0x0" ::: "memory");
#endif
#endif
}

// stage WhT[b][0:4][0:128][m0:m0+32] (32 KB) into one LDS buffer (whole workgroup)
__device__ __forceinline__ void stage_chunk(const __bf16* __restrict__ WhT, int b, int m0,
                                            __bf16* buf, int tid) {
#pragma unroll
  for (int t = 0; t < 2; ++t) {
    int rr = tid + t * 256;                 // 512 (h,o) rows of 64B
    int hh = rr >> 7, oo = rr & 127;
    const char* src = (const char*)(WhT + ((size_t)(b * 4 + hh) * OUT_F + oo) * NNODES + m0);
    char* dst = (char*)&buf[rr * 32];
#pragma unroll
    for (int q = 0; q < 4; ++q)
      copy16_to_lds(src + q * 16, dst + q * 16);
  }
}

__global__ void __launch_bounds__(256) attn_kernel(const float* __restrict__ s1,
                                                   const float* __restrict__ s2,
                                                   const float* __restrict__ rowC,
                                                   const int* __restrict__ adj,
                                                   const __bf16* __restrict__ WhT,
                                                   float* __restrict__ out) {
  // double-buffered Wh chunk, [buf][h][o][m32] : 2 x 32 KB
  __shared__ __align__(32) __bf16 Bbuf[2][HEADS * OUT_F * 32];

  int b = blockIdx.x >> 4, grp = blockIdx.x & 15;
  int tid = threadIdx.x;
  int wave = tid >> 5, lane = tid & 31, lo = lane & 15, hi = lane >> 4;
  int n0 = ((grp << 3) + wave) << 4;
  int n  = n0 + lo;

  float s1v[4], cv[4];
#pragma unroll
  for (int hh = 0; hh < 4; ++hh) {
    int r = ((b * 4 + hh) << 11) + n;
    s1v[hh] = s1[r];
    cv[hh]  = rowC[r];
  }
  const int* adjRow = adj + (size_t)n * NNODES;
  v8f acc[8] = {};

  // prologue: stage chunk 0
  stage_chunk(WhT, b, 0, Bbuf[0], tid);
  async_wait_all();
  __syncthreads();

  int cur = 0;
  for (int m0 = 0; m0 < NNODES; m0 += 32) {
    // kick off next chunk's async copies into the other buffer (overlaps with WMMA below)
    if (m0 + 32 < NNODES)
      stage_chunk(WhT, b, m0 + 32, Bbuf[cur ^ 1], tid);

    const int* ap = adjRow + m0 + 16 * hi;     // adj in A-fragment lane layout
    int4 ad0 = *(const int4*)(ap);
    int4 ad1 = *(const int4*)(ap + 4);
    int4 ad2 = *(const int4*)(ap + 8);
    int4 ad3 = *(const int4*)(ap + 12);

#pragma unroll
    for (int hh = 0; hh < 4; ++hh) {
      const float* sp = s2 + (((b * 4 + hh) << 11) + m0 + 16 * hi);
      float4 f0 = *(const float4*)(sp);
      float4 f1 = *(const float4*)(sp + 4);
      float4 f2 = *(const float4*)(sp + 8);
      float4 f3 = *(const float4*)(sp + 12);
      float sv = s1v[hh], c = cv[hh];
      v16bf A;
      A[0]  = pval(sv, c, ad0.x, f0.x); A[1]  = pval(sv, c, ad0.y, f0.y);
      A[2]  = pval(sv, c, ad0.z, f0.z); A[3]  = pval(sv, c, ad0.w, f0.w);
      A[4]  = pval(sv, c, ad1.x, f1.x); A[5]  = pval(sv, c, ad1.y, f1.y);
      A[6]  = pval(sv, c, ad1.z, f1.z); A[7]  = pval(sv, c, ad1.w, f1.w);
      A[8]  = pval(sv, c, ad2.x, f2.x); A[9]  = pval(sv, c, ad2.y, f2.y);
      A[10] = pval(sv, c, ad2.z, f2.z); A[11] = pval(sv, c, ad2.w, f2.w);
      A[12] = pval(sv, c, ad3.x, f3.x); A[13] = pval(sv, c, ad3.y, f3.y);
      A[14] = pval(sv, c, ad3.z, f3.z); A[15] = pval(sv, c, ad3.w, f3.w);
#pragma unroll
      for (int j = 0; j < 8; ++j) {
        const v16bf* Bp = (const v16bf*)&Bbuf[cur][(hh * OUT_F + (j << 4) + lo) * 32 + 16 * hi];
        acc[j] = __builtin_amdgcn_wmma_f32_16x16x32_bf16(false, A, false, *Bp,
                                                         (short)0, acc[j], false, false);
      }
    }

    async_wait_all();   // our wave's async copies for the next chunk are done
    __syncthreads();    // whole workgroup done reading cur / writing next
    cur ^= 1;
  }

  // epilogue: mean over heads (x0.25) + ELU, write (b, n, o) f32
#pragma unroll
  for (int j = 0; j < 8; ++j) {
    int o = (j << 4) + lo;
#pragma unroll
    for (int r = 0; r < 8; ++r) {
      float v = 0.25f * acc[j][r];
      v = v > 0.f ? v : (__expf(v) - 1.f);
      out[(size_t)(b * NNODES + n0 + 8 * hi + r) * OUT_F + o] = v;
    }
  }
}

// ---------------- launcher ----------------
extern "C" void kernel_launch(void* const* d_in, const int* in_sizes, int n_in,
                              void* d_out, int out_size, void* d_ws, size_t ws_size,
                              hipStream_t stream) {
  (void)in_sizes; (void)n_in; (void)out_size; (void)ws_size;
  const float* h   = (const float*)d_in[0];
  const int*   adj = (const int*)d_in[1];
  const float* W   = (const float*)d_in[2];
  const float* a   = (const float*)d_in[3];
  float* out = (float*)d_out;
  char* ws = (char*)d_ws;

  // workspace layout (~25 MB total; all intermediates stay L2-resident)
  __bf16* hB   = (__bf16*)(ws);                       //  8,388,608 B  h in bf16
  __bf16* WtB  = (__bf16*)(ws + 8388608);             //    262,144 B  W^T bf16 [h][o][k]
  __bf16* WhT  = (__bf16*)(ws + 8650752);             // 16,777,216 B  Wh^T bf16 [b][h][o][n]
  float*  s1   = (float*)(ws + 25427968);             //    262,144 B
  float*  s2   = (float*)(ws + 25690112);             //    262,144 B
  float*  rowC = (float*)(ws + 25952256);             //    262,144 B

  cvt_h_kernel<<<2048, 256, 0, stream>>>(h, hB, NB * NNODES * IN_F);
  cvt_w_kernel<<<512, 256, 0, stream>>>(W, WtB);
  gemm1_kernel<<<512, 256, 0, stream>>>(hB, WtB, WhT);
  s12_kernel<<<256, 256, 0, stream>>>(WhT, a, s1, s2);
  stats_kernel<<<8192, 256, 0, stream>>>(s1, s2, adj, rowC);
  attn_kernel<<<128, 256, 0, stream>>>(s1, s2, rowC, adj, WhT, out);
}
